// JetTaggerNetwork_57234734186738
// MI455X (gfx1250) — matrix operand order repaired and checked by reading, not compiled, not run on hardware
//
#include <hip/hip_runtime.h>

typedef __attribute__((ext_vector_type(16))) _Float16     v16h;
typedef __attribute__((ext_vector_type(8)))  float        v8f;
typedef __attribute__((ext_vector_type(4)))  unsigned int v4u;
typedef __attribute__((ext_vector_type(8)))  unsigned int v8u;

#define WAVES_PER_BLOCK 8

// ---------------------------------------------------------------------------
// Zero the output accumulators (segment_sum targets).
// ---------------------------------------------------------------------------
__global__ void zero_kernel(float* __restrict__ out, long long n) {
  long long i = (long long)blockIdx.x * blockDim.x + threadIdx.x;
  long long stride = (long long)gridDim.x * blockDim.x;
  for (; i < n; i += stride) out[i] = 0.0f;
}

// ---------------------------------------------------------------------------
// WMMA helper (CDNA5 wave32, 16x16x32 f16 -> f32)
// ---------------------------------------------------------------------------
__device__ __forceinline__ v8f wmma_f16(v16h a, v16h b, v8f c) {
  return __builtin_amdgcn_wmma_f32_16x16x32_f16(false, a, false, b, (short)0, c,
                                                false, false);
}

__device__ __forceinline__ v16h pack16(v4u lo, v4u hi) {
  v8u t;
#pragma unroll
  for (int i = 0; i < 4; ++i) { t[i] = lo[i]; t[i + 4] = hi[i]; }
  return __builtin_bit_cast(v16h, t);
}

// B fragment from LDS-resident transposed f16 weights Wt[N][Kpad].
// ISA layout: lane = 16*g + n holds halfs [8g,8g+8) and [16+8g,16+8g+8)
// of the K step starting at kbase -> two aligned ds_load_b128.
__device__ __forceinline__ v16h b_from_lds(const _Float16* __restrict__ Wt,
                                           int Kpad, int nbase, int kbase) {
  int lane = threadIdx.x & 31;
  int g = lane >> 4, n = lane & 15;
  const v4u* p0 = (const v4u*)&Wt[(nbase + n) * Kpad + kbase + 8 * g];
  const v4u* p1 = (const v4u*)&Wt[(nbase + n) * Kpad + kbase + 16 + 8 * g];
  return pack16(p0[0], p1[0]);
}

// A fragment from LDS-resident row-major f16 activations H[16][64].
// ISA layout: lane = 16*g + m; halfs [8g,8g+8) and [16+8g,16+8g+8) of the
// selected 32-wide K step -> two aligned ds_load_b128.
__device__ __forceinline__ v16h a_from_lds(const _Float16* __restrict__ H, int ks) {
  int lane = threadIdx.x & 31;
  int g = lane >> 4, m = lane & 15;
  const v4u* p0 = (const v4u*)&H[m * 64 + 32 * ks + 8 * g];
  const v4u* p1 = (const v4u*)&H[m * 64 + 32 * ks + 16 + 8 * g];
  return pack16(p0[0], p1[0]);
}

// ---------------------------------------------------------------------------
// MLP [K0 -> 64 -> 64 -> 32], ReLU on hidden layers.
// Weights staged once per block into LDS (transposed, f16, K0 padded to 32).
// Persistent waves: each wave grid-strides over 16-row tiles; activations
// staged through a per-wave LDS tile (in-order LDS within a wave, no barrier).
// ---------------------------------------------------------------------------
template <int K0>
__global__ void __launch_bounds__(256)
mlp_wmma_kernel(const float* __restrict__ X,
                const float* __restrict__ W1, const float* __restrict__ B1,
                const float* __restrict__ W2, const float* __restrict__ B2,
                const float* __restrict__ W3, const float* __restrict__ B3,
                float* __restrict__ rep, int nrows) {
  __shared__ __align__(16) _Float16 WT1[64 * 32];             // [n][k], k padded
  __shared__ __align__(16) _Float16 WT2[64 * 64];             // [n][k]
  __shared__ __align__(16) _Float16 WT3[32 * 64];             // [n][k]
  __shared__ __align__(16) _Float16 Hs[WAVES_PER_BLOCK][16 * 64];

  const int tid = threadIdx.x;

  // ---- one-time cooperative weight staging (f32 -> f16, transposed) ----
#pragma unroll 1
  for (int i = tid; i < 64 * 32; i += 256) {
    int n = i >> 5, k = i & 31;
    WT1[i] = (k < K0) ? (_Float16)W1[k * 64 + n] : (_Float16)0.0f;
  }
#pragma unroll 1
  for (int i = tid; i < 64 * 64; i += 256) {
    int n = i >> 6, k = i & 63;
    WT2[i] = (_Float16)W2[k * 64 + n];
  }
#pragma unroll 1
  for (int i = tid; i < 32 * 64; i += 256) {
    int n = i >> 6, k = i & 63;
    WT3[i] = (_Float16)W3[k * 32 + n];
  }
  __syncthreads();

  const int wave = tid >> 5;
  const int lane = tid & 31;
  const int g = lane >> 4;
  const int mn = lane & 15;  // row (A-role) or column (B/C-role) within a tile
  _Float16* H = Hs[wave];

  // ---- biases are tile-invariant: preload per lane ----
  float bias1[4], bias2[4], bias3[2];
#pragma unroll
  for (int jn = 0; jn < 4; ++jn) bias1[jn] = B1[jn * 16 + mn];
#pragma unroll
  for (int jn = 0; jn < 4; ++jn) bias2[jn] = B2[jn * 16 + mn];
#pragma unroll
  for (int jn = 0; jn < 2; ++jn) bias3[jn] = B3[jn * 16 + mn];

  const long long tiles = ((long long)nrows + 15) >> 4;
  const long long wstride = (long long)gridDim.x * WAVES_PER_BLOCK;

  for (long long tile = (long long)blockIdx.x * WAVES_PER_BLOCK + wave;
       tile < tiles; tile += wstride) {
    const long long base = tile * 16;

    // ----- Layer 1: A = X tile (16 x K0, zero-padded to K=32) -----
    long long arow = base + mn;
    if (arow > (long long)nrows - 1) arow = nrows - 1;  // clamp; stores guarded
    const float* xr = X + (size_t)arow * K0;
    v16h a1;
#pragma unroll
    for (int v = 0; v < 8; ++v) {
      int k = (v < 4) ? (8 * g + 2 * v) : (16 + 8 * g + 2 * (v - 4));
      // branch-free: unconditional clamped load + select (no exec-mask chains)
      float x0 = xr[(k < K0) ? k : 0];
      float x1 = xr[(k + 1 < K0) ? (k + 1) : 0];
      a1[2 * v]     = (k < K0)     ? (_Float16)x0 : (_Float16)0.0f;
      a1[2 * v + 1] = (k + 1 < K0) ? (_Float16)x1 : (_Float16)0.0f;
    }
#pragma unroll
    for (int jn = 0; jn < 4; ++jn) {
      v8f acc = {};
      acc = wmma_f16(a1, b_from_lds(WT1, 32, jn * 16, 0), acc);
#pragma unroll
      for (int r = 0; r < 8; ++r) {
        float h = acc[r] + bias1[jn];
        h = h > 0.0f ? h : 0.0f;  // ReLU
        H[(8 * g + r) * 64 + jn * 16 + mn] = (_Float16)h;
      }
    }

    // ----- Layer 2: 64 -> 64 (read both A frags before overwriting H) -----
    v16h a20 = a_from_lds(H, 0);
    v16h a21 = a_from_lds(H, 1);
    v8f c2[4];
#pragma unroll
    for (int jn = 0; jn < 4; ++jn) {
      v8f acc = {};
      acc = wmma_f16(a20, b_from_lds(WT2, 64, jn * 16, 0), acc);
      acc = wmma_f16(a21, b_from_lds(WT2, 64, jn * 16, 32), acc);
      c2[jn] = acc;
    }
#pragma unroll
    for (int jn = 0; jn < 4; ++jn) {
#pragma unroll
      for (int r = 0; r < 8; ++r) {
        float h = c2[jn][r] + bias2[jn];
        h = h > 0.0f ? h : 0.0f;
        H[(8 * g + r) * 64 + jn * 16 + mn] = (_Float16)h;
      }
    }

    // ----- Layer 3: 64 -> 32, no activation, store f32 reps -----
    v16h a30 = a_from_lds(H, 0);
    v16h a31 = a_from_lds(H, 1);
#pragma unroll
    for (int jn = 0; jn < 2; ++jn) {
      v8f acc = {};
      acc = wmma_f16(a30, b_from_lds(WT3, 64, jn * 16, 0), acc);
      acc = wmma_f16(a31, b_from_lds(WT3, 64, jn * 16, 32), acc);
#pragma unroll
      for (int r = 0; r < 8; ++r) {
        long long row = base + 8 * g + r;
        if (row < nrows)
          rep[(size_t)row * 32 + jn * 16 + mn] = acc[r] + bias3[jn];
      }
    }
  }
}

// ---------------------------------------------------------------------------
// Edge scatter: one edge per wave32. Lane f handles rep feature f (coalesced
// 128B atomic burst per row); lanes 0-4 add the type embedding, lanes 0-9 the
// common variables (tracks/leptons only).
// ---------------------------------------------------------------------------
__global__ void scatter_kernel(const float* __restrict__ rep,
                               const int* __restrict__ type,
                               const float* __restrict__ embed,   // [5][5]
                               const float* __restrict__ common,  // may be null
                               const int* __restrict__ src,
                               const int* __restrict__ dst,
                               long long n_edges,
                               float* __restrict__ out_emb,
                               float* __restrict__ out_feat,
                               float* __restrict__ out_common) {
  long long e = ((long long)blockIdx.x * blockDim.x + threadIdx.x) >> 5;
  int lane = threadIdx.x & 31;
  if (e >= n_edges) return;
  int s = src[e];
  int d = dst[e];
  atomicAdd(&out_feat[(size_t)d * 32 + lane], rep[(size_t)s * 32 + lane]);
  if (lane < 5) {
    int t = type[s];
    atomicAdd(&out_emb[(size_t)d * 5 + lane], embed[t * 5 + lane]);
  }
  if (common != nullptr && lane < 10) {
    atomicAdd(&out_common[(size_t)d * 10 + lane], common[(size_t)s * 10 + lane]);
  }
}

// ---------------------------------------------------------------------------
// Launcher
// ---------------------------------------------------------------------------
static inline int mlp_blocks(long long nrows) {
  long long tiles = (nrows + 15) >> 4;
  long long blocks = (tiles + WAVES_PER_BLOCK - 1) / WAVES_PER_BLOCK;
  if (blocks > 1024) blocks = 1024;  // persistent waves grid-stride the rest
  return (int)blocks;
}

extern "C" void kernel_launch(void* const* d_in, const int* in_sizes, int n_in,
                              void* d_out, int out_size, void* d_ws, size_t ws_size,
                              hipStream_t stream) {
  (void)n_in; (void)ws_size;
  const float* track_vars   = (const float*)d_in[0];
  const float* lep_vars     = (const float*)d_in[1];
  const float* cell_vars    = (const float*)d_in[2];
  const float* track_common = (const float*)d_in[3];
  const float* lep_common   = (const float*)d_in[4];
  const int*   track_type   = (const int*)d_in[5];
  const int*   lep_type     = (const int*)d_in[6];
  const int*   cell_type    = (const int*)d_in[7];
  const int*   t2n_src = (const int*)d_in[8];
  const int*   t2n_dst = (const int*)d_in[9];
  const int*   l2n_src = (const int*)d_in[10];
  const int*   l2n_dst = (const int*)d_in[11];
  const int*   c2n_src = (const int*)d_in[12];
  const int*   c2n_dst = (const int*)d_in[13];
  const float* embed   = (const float*)d_in[14];
  // MLP params flatten in pytree order: (W1,b1),(W2,b2),(W3,b3) per entity.
  const float* tW1 = (const float*)d_in[15]; const float* tB1 = (const float*)d_in[16];
  const float* tW2 = (const float*)d_in[17]; const float* tB2 = (const float*)d_in[18];
  const float* tW3 = (const float*)d_in[19]; const float* tB3 = (const float*)d_in[20];
  const float* lW1 = (const float*)d_in[21]; const float* lB1 = (const float*)d_in[22];
  const float* lW2 = (const float*)d_in[23]; const float* lB2 = (const float*)d_in[24];
  const float* lW3 = (const float*)d_in[25]; const float* lB3 = (const float*)d_in[26];
  const float* cW1 = (const float*)d_in[27]; const float* cB1 = (const float*)d_in[28];
  const float* cW2 = (const float*)d_in[29]; const float* cB2 = (const float*)d_in[30];
  const float* cW3 = (const float*)d_in[31]; const float* cB3 = (const float*)d_in[32];

  const int NT = in_sizes[5];
  const int NL = in_sizes[6];
  const int NC = in_sizes[7];
  const long long ET = in_sizes[8];
  const long long EL = in_sizes[10];
  const long long EC = in_sizes[12];

  // Output layout: [N,5] emb | [N,32] feat | [N,10] common, concatenated flat.
  const long long N = (long long)out_size / 47;
  float* out_emb    = (float*)d_out;
  float* out_feat   = out_emb + N * 5;
  float* out_common = out_feat + N * 32;

  // Workspace: f32 reps per entity.
  float* track_rep = (float*)d_ws;
  float* lep_rep   = track_rep + (size_t)NT * 32;
  float* cell_rep  = lep_rep + (size_t)NL * 32;

  // 1) zero accumulators
  zero_kernel<<<4096, 256, 0, stream>>>((float*)d_out, (long long)out_size);

  // 2) MLP reps (WMMA)
  mlp_wmma_kernel<18><<<mlp_blocks(NT), 256, 0, stream>>>(
      track_vars, tW1, tB1, tW2, tB2, tW3, tB3, track_rep, NT);
  mlp_wmma_kernel<14><<<mlp_blocks(NL), 256, 0, stream>>>(
      lep_vars, lW1, lB1, lW2, lB2, lW3, lB3, lep_rep, NL);
  mlp_wmma_kernel<8><<<mlp_blocks(NC), 256, 0, stream>>>(
      cell_vars, cW1, cB1, cW2, cB2, cW3, cB3, cell_rep, NC);

  // 3) edge scatters (one edge per wave)
  {
    long long threads = ET * 32;
    scatter_kernel<<<(int)((threads + 255) / 256), 256, 0, stream>>>(
        track_rep, track_type, embed, track_common, t2n_src, t2n_dst, ET,
        out_emb, out_feat, out_common);
  }
  {
    long long threads = EL * 32;
    scatter_kernel<<<(int)((threads + 255) / 256), 256, 0, stream>>>(
        lep_rep, lep_type, embed, lep_common, l2n_src, l2n_dst, EL,
        out_emb, out_feat, out_common);
  }
  {
    long long threads = EC * 32;
    scatter_kernel<<<(int)((threads + 255) / 256), 256, 0, stream>>>(
        cell_rep, cell_type, embed, nullptr, c2n_src, c2n_dst, EC,
        out_emb, out_feat, out_common);
  }
}
